// SentenceEmbedding_3161095930187
// MI455X (gfx1250) — matrix operand order repaired and checked by reading, not compiled, not run on hardware
//
#include <hip/hip_runtime.h>

typedef __attribute__((ext_vector_type(16))) _Float16 v16h;
typedef __attribute__((ext_vector_type(8)))  _Float16 v8h;
typedef __attribute__((ext_vector_type(8)))  float    v8f;

#define B_    512
#define L_    128
#define WD_   300
#define PD_   50
#define C_    256
#define KH_   3
#define D_    400      // WD + 2*PD
#define DP_   416      // D padded to multiple of 32 (13*32)
#define LC_   126      // L - KH + 1
#define KTOT_ (KH_*DP_)    // 1248
#define KSTEPS_ (KTOT_/32) // 39
#define M_    (B_*LC_)     // 64512 (divisible by 32)

// ---------------------------------------------------------------------------
// Zero-init the pooled output (harness poisons d_out; atomic-max needs 0.0)
// ---------------------------------------------------------------------------
__global__ void zero_out_kernel(float* __restrict__ o, int n) {
  int i = blockIdx.x * blockDim.x + threadIdx.x;
  if (i < n) o[i] = 0.0f;
}

// ---------------------------------------------------------------------------
// Gather embeddings -> padded f16 activation X[b][l][0..415]
// ---------------------------------------------------------------------------
__global__ void gather_embed_kernel(const int* __restrict__ sent,
                                    const int* __restrict__ pos1,
                                    const int* __restrict__ pos2,
                                    const float* __restrict__ Wword,
                                    const float* __restrict__ Wp1,
                                    const float* __restrict__ Wp2,
                                    _Float16* __restrict__ X) {
  const int bl = blockIdx.x;              // 0 .. B*L-1
  const int si = sent[bl];
  const int p1 = pos1[bl];
  const int p2 = pos2[bl];
  _Float16* row = X + (size_t)bl * DP_;
  for (int d = threadIdx.x; d < DP_; d += blockDim.x) {
    float v;
    if (d < WD_)            v = Wword[(size_t)si * WD_ + d];
    else if (d < WD_ + PD_) v = Wp1[p1 * PD_ + (d - WD_)];
    else if (d < D_)        v = Wp2[p2 * PD_ + (d - WD_ - PD_)];
    else                    v = 0.0f;     // K padding
    row[d] = (_Float16)v;
  }
}

// ---------------------------------------------------------------------------
// Transpose/pad conv weights -> Wt[c][kh*416 + d] in f16 (B-matrix source)
// ---------------------------------------------------------------------------
__global__ void prep_weights_kernel(const float* __restrict__ cw,
                                    _Float16* __restrict__ Wt) {
  const int c = blockIdx.x;               // 0 .. C-1
  for (int k = threadIdx.x; k < KTOT_; k += blockDim.x) {
    const int kh = k / DP_;
    const int d  = k - kh * DP_;
    float v = (d < D_) ? cw[(size_t)c * (KH_ * D_) + kh * D_ + d] : 0.0f;
    Wt[(size_t)c * KTOT_ + k] = (_Float16)v;
  }
}

// ---------------------------------------------------------------------------
// Conv-as-GEMM with WMMA + fused bias/ReLU/piecewise-max epilogue.
// Block = 128 threads = 4 waves. Wave w computes a 32M x 64N tile
// (2 A fragments x 4 B fragments = 8 f32 accumulators); the 4 waves of a
// block cover all 256 channels. grid.x = M/32 = 2016.
// ---------------------------------------------------------------------------
__global__ __launch_bounds__(128) void conv_wmma_kernel(
    const _Float16* __restrict__ X,     // [B*L][DP_]
    const _Float16* __restrict__ Wt,    // [C][KTOT_]
    const float*    __restrict__ bias,  // [C]
    const int*      __restrict__ widx,  // [B][2]
    float*          __restrict__ out)   // [B][C*3], pre-zeroed
{
  const int lane  = threadIdx.x & 31;
  const int wave  = threadIdx.x >> 5;          // 0..3 -> N base = wave*64
  const int mbase = blockIdx.x * 32;           // 32 rows per block
  const int khalf = lane >> 4;                 // 0/1: K half-group

  // --- A-side: this lane supplies rows r0 = mbase+(lane&15) and r1 = r0+16.
  const int r0 = mbase + (lane & 15);
  const int r1 = r0 + 16;
  const int bA0 = r0 / LC_, lA0 = r0 - bA0 * LC_;
  const int bA1 = r1 / LC_, lA1 = r1 - bA1 * LC_;
  // KH=3 window of row l is contiguous: 1248 halves at (b*128 + l)*416
  const _Float16* aBase0 = X + ((size_t)bA0 * L_ + lA0) * DP_;
  const _Float16* aBase1 = X + ((size_t)bA1 * L_ + lA1) * DP_;

  // --- B-side: lane supplies column n = wave*64 + j*16 + (lane&15)
  const int ncol = wave * 64 + (lane & 15);
  const _Float16* b0p = Wt + (size_t)(ncol     ) * KTOT_;
  const _Float16* b1p = Wt + (size_t)(ncol + 16) * KTOT_;
  const _Float16* b2p = Wt + (size_t)(ncol + 32) * KTOT_;
  const _Float16* b3p = Wt + (size_t)(ncol + 48) * KTOT_;

  v8f a0c0 = {}, a0c1 = {}, a0c2 = {}, a0c3 = {};
  v8f a1c0 = {}, a1c1 = {}, a1c2 = {}, a1c3 = {};

  for (int kk = 0; kk < KSTEPS_; ++kk) {
    const int k0 = kk * 32;

    // A fragments: lane-group khalf holds contiguous
    // K = k0+khalf*16 .. +15 (matching B).
    union { v16h v; v8h h[2]; } a0, a1;
    a0.h[0] = *(const v8h*)(aBase0 + k0 + khalf * 16);
    a0.h[1] = *(const v8h*)(aBase0 + k0 + khalf * 16 + 8);
    a1.h[0] = *(const v8h*)(aBase1 + k0 + khalf * 16);
    a1.h[1] = *(const v8h*)(aBase1 + k0 + khalf * 16 + 8);

    // B fragments: lane-group khalf holds contiguous K = k0+khalf*16 .. +15.
    const int kb = k0 + khalf * 16;
    v16h b0 = *(const v16h*)(b0p + kb);
    v16h b1 = *(const v16h*)(b1p + kb);
    v16h b2 = *(const v16h*)(b2p + kb);
    v16h b3 = *(const v16h*)(b3p + kb);

    // Speculative prefetch of the B stream ~8 k-steps ahead (gfx1250
    // global_prefetch; OOB speculative prefetches are silently dropped).
    __builtin_prefetch(b0p + kb + 256, 0, 1);
    __builtin_prefetch(b2p + kb + 256, 0, 1);

    a0c0 = __builtin_amdgcn_wmma_f32_16x16x32_f16(false, a0.v, false, b0,
                                                  (short)0, a0c0, false, false);
    a0c1 = __builtin_amdgcn_wmma_f32_16x16x32_f16(false, a0.v, false, b1,
                                                  (short)0, a0c1, false, false);
    a0c2 = __builtin_amdgcn_wmma_f32_16x16x32_f16(false, a0.v, false, b2,
                                                  (short)0, a0c2, false, false);
    a0c3 = __builtin_amdgcn_wmma_f32_16x16x32_f16(false, a0.v, false, b3,
                                                  (short)0, a0c3, false, false);
    a1c0 = __builtin_amdgcn_wmma_f32_16x16x32_f16(false, a1.v, false, b0,
                                                  (short)0, a1c0, false, false);
    a1c1 = __builtin_amdgcn_wmma_f32_16x16x32_f16(false, a1.v, false, b1,
                                                  (short)0, a1c1, false, false);
    a1c2 = __builtin_amdgcn_wmma_f32_16x16x32_f16(false, a1.v, false, b2,
                                                  (short)0, a1c2, false, false);
    a1c3 = __builtin_amdgcn_wmma_f32_16x16x32_f16(false, a1.v, false, b3,
                                                  (short)0, a1c3, false, false);
  }

  // --- Epilogue: C/D layout -> element (vreg v, lane): row = v + khalf*8,
  // col = lane&15 within each 16x16 tile. relu(acc+bias) >= 0, and nonneg
  // f32 bit patterns order like signed ints -> integer atomic max pools
  // correctly into the zero-initialized output (empty segments stay 0.0).
  const float bv0 = bias[ncol];
  const float bv1 = bias[ncol + 16];
  const float bv2 = bias[ncol + 32];
  const float bv3 = bias[ncol + 48];
  int* o = (int*)out;

#pragma unroll
  for (int t = 0; t < 2; ++t) {
    const v8f c0 = t ? a1c0 : a0c0;
    const v8f c1 = t ? a1c1 : a0c1;
    const v8f c2 = t ? a1c2 : a0c2;
    const v8f c3 = t ? a1c3 : a0c3;
#pragma unroll
    for (int v = 0; v < 8; ++v) {
      const int m = mbase + t * 16 + v + khalf * 8;
      const int b = m / LC_;
      const int l = m - b * LC_;
      const int w0 = widx[b * 2 + 0];
      const int w1 = widx[b * 2 + 1];
      const int seg = (l < w0) ? 0 : ((l < w1 - 1) ? 1 : 2);
      const int base = b * (3 * C_) + seg;   // out[b][c*3 + seg]
      float r;
      r = fmaxf(c0[v] + bv0, 0.0f); atomicMax(o + base + (ncol     ) * 3, __float_as_int(r));
      r = fmaxf(c1[v] + bv1, 0.0f); atomicMax(o + base + (ncol + 16) * 3, __float_as_int(r));
      r = fmaxf(c2[v] + bv2, 0.0f); atomicMax(o + base + (ncol + 32) * 3, __float_as_int(r));
      r = fmaxf(c3[v] + bv3, 0.0f); atomicMax(o + base + (ncol + 48) * 3, __float_as_int(r));
    }
  }
}

// ---------------------------------------------------------------------------
extern "C" void kernel_launch(void* const* d_in, const int* in_sizes, int n_in,
                              void* d_out, int out_size, void* d_ws, size_t ws_size,
                              hipStream_t stream) {
  const int*   sent  = (const int*)d_in[0];
  const int*   pos1  = (const int*)d_in[1];
  const int*   pos2  = (const int*)d_in[2];
  const int*   widx  = (const int*)d_in[3];
  const float* Wword = (const float*)d_in[4];
  const float* Wp1   = (const float*)d_in[5];
  const float* Wp2   = (const float*)d_in[6];
  const float* convw = (const float*)d_in[7];
  const float* convb = (const float*)d_in[8];
  float*       out   = (float*)d_out;

  // Workspace layout: X f16 (B*L*416) then Wt f16 (C*1248); ~55.2 MB total.
  _Float16* X  = (_Float16*)d_ws;
  _Float16* Wt = (_Float16*)((char*)d_ws + (size_t)B_ * L_ * DP_ * sizeof(_Float16));

  const int outN = B_ * 3 * C_;
  zero_out_kernel<<<(outN + 255) / 256, 256, 0, stream>>>(out, outN);
  gather_embed_kernel<<<B_ * L_, 128, 0, stream>>>(sent, pos1, pos2,
                                                   Wword, Wp1, Wp2, X);
  prep_weights_kernel<<<C_, 128, 0, stream>>>(convw, Wt);
  conv_wmma_kernel<<<M_ / 32, 128, 0, stream>>>(X, Wt, convb, widx, out);
}
